// GPTAttention_45681272160778
// MI455X (gfx1250) — 16-bit and 8-bit matrix operand order repaired and checked by reading, compiled, not run on hardware
//
#include <hip/hip_runtime.h>
#include <math.h>


#define B_  2
#define T_  2048
#define C_  1024
#define H_  16
#define D_  64
#define BT  (B_ * T_)
#define C3  (3 * C_)

typedef __attribute__((ext_vector_type(16))) _Float16     v16h;
typedef __attribute__((ext_vector_type(8)))  float        v8f;
typedef __attribute__((ext_vector_type(8)))  int          v8i;
typedef __attribute__((ext_vector_type(4)))  int          i32x4;
typedef __attribute__((ext_vector_type(2)))  unsigned int u32x2;
typedef __attribute__((ext_vector_type(4)))  unsigned int u32x4;

union AFrag8 { v8i v; u32x2 d[4]; };
union BFrag8 { v8i v; u32x4 q[2]; };
union HFrag  { v16h v; u32x4 q[2]; };

#if defined(__has_builtin)
#if __has_builtin(__builtin_amdgcn_tensor_load_to_lds)
#define HAVE_TDM 1
#endif
#if __has_builtin(__builtin_amdgcn_s_wait_tensorcnt)
#define HAVE_TCNT 1
#endif
#endif

static constexpr size_t OFF_AMAX  = 0;
static constexpr size_t OFF_XQ    = 256;
static constexpr size_t OFF_WQKV  = OFF_XQ    + (size_t)BT * C_;
static constexpr size_t OFF_WPROJ = OFF_WQKV  + (size_t)C3 * C_;
static constexpr size_t OFF_QKV   = OFF_WPROJ + (size_t)C_ * C_;
static constexpr size_t OFF_AOUT  = OFF_QKV   + (size_t)3 * BT * C_ * 2;
static constexpr size_t OFF_OUTQ  = OFF_AOUT  + (size_t)BT * C_ * 4;

__device__ __forceinline__ void wave_lds_sync() {
  __builtin_amdgcn_wave_barrier();
  asm volatile("s_wait_dscnt 0" ::: "memory");
  __builtin_amdgcn_wave_barrier();
}

__device__ __forceinline__ void wait_tensor0() {
#ifdef HAVE_TCNT
  __builtin_amdgcn_s_wait_tensorcnt(0);
#else
  asm volatile("s_wait_tensorcnt 0" ::: "memory");
#endif
}

__global__ void init_amax_kernel(float* amax) {
  if (threadIdx.x < 4) amax[threadIdx.x] = 0.0f;
}

__global__ void amax_kernel(const float* __restrict__ x, int n4, float* slot) {
  int i = blockIdx.x * blockDim.x + threadIdx.x;
  int stride = gridDim.x * blockDim.x;
  float m = 0.0f;
  const float4* xv = (const float4*)x;
  for (int j = i; j < n4; j += stride) {
    float4 v = xv[j];
    m = fmaxf(m, fmaxf(fmaxf(fabsf(v.x), fabsf(v.y)),
                       fmaxf(fabsf(v.z), fabsf(v.w))));
  }
  #pragma unroll
  for (int mk = 1; mk < 32; mk <<= 1) m = fmaxf(m, __shfl_xor(m, mk, 32));
  if ((threadIdx.x & 31) == 0) atomicMax((unsigned int*)slot, __float_as_uint(m));
}

__global__ void quant_kernel(const float* __restrict__ x, int n4,
                             const float* __restrict__ amax_slot,
                             signed char* __restrict__ q) {
  float scale = fmaxf(amax_slot[0] * (1.0f / 127.0f), 1e-8f);
  float inv = 1.0f / scale;
  int i = blockIdx.x * blockDim.x + threadIdx.x;
  int stride = gridDim.x * blockDim.x;
  const float4* xv = (const float4*)x;
  for (int j = i; j < n4; j += stride) {
    float4 v = xv[j];
    signed char o[4];
    o[0] = (signed char)(int)fminf(fmaxf(rintf(v.x * inv), -127.0f), 127.0f);
    o[1] = (signed char)(int)fminf(fmaxf(rintf(v.y * inv), -127.0f), 127.0f);
    o[2] = (signed char)(int)fminf(fmaxf(rintf(v.z * inv), -127.0f), 127.0f);
    o[3] = (signed char)(int)fminf(fmaxf(rintf(v.w * inv), -127.0f), 127.0f);
    *(int*)(q + (size_t)j * 4) = *(int*)o;
  }
}

__global__ __launch_bounds__(256) void
gemm_iu8_kernel(const signed char* __restrict__ A,
                const signed char* __restrict__ W,
                const float* __restrict__ bias,
                const float* __restrict__ amax, int ai, int wi,
                int M, int N, int K, int mode, void* out_p) {
  int wid  = (blockIdx.x * blockDim.x + threadIdx.x) >> 5;
  int lane = threadIdx.x & 31;
  int nStrips = N >> 6;
  int mTile = (wid / nStrips) << 4;
  int nBase = (wid % nStrips) << 6;
  if (mTile >= M) return;
  int hl = lane >> 4;
  int l16 = lane & 15;

  v8i acc[4];
  #pragma unroll
  for (int j = 0; j < 4; ++j)
    #pragma unroll
    for (int r = 0; r < 8; ++r) acc[j][r] = 0;

  const signed char* arow = A + (size_t)(mTile + l16) * K;
  const signed char* wrow[4];
  #pragma unroll
  for (int j = 0; j < 4; ++j)
    wrow[j] = W + (size_t)(nBase + j * 16 + l16) * K;

  for (int k0 = 0; k0 < K; k0 += 64) {
    __builtin_prefetch(arow + k0 + 256, 0, 3);
    __builtin_prefetch(wrow[0] + k0 + 256, 0, 3);
    __builtin_prefetch(wrow[2] + k0 + 256, 0, 3);
    AFrag8 a;
    #pragma unroll
    for (int c = 0; c < 4; ++c)
      a.d[c] = *(const u32x2*)(arow + k0 + (c >> 1) * 32 + hl * 16 + (c & 1) * 8);
    #pragma unroll
    for (int j = 0; j < 4; ++j) {
      BFrag8 b;
      b.q[0] = *(const u32x4*)(wrow[j] + k0 + hl * 16);
      b.q[1] = *(const u32x4*)(wrow[j] + k0 + 32 + hl * 16);
      acc[j] = __builtin_amdgcn_wmma_i32_16x16x64_iu8(true, a.v, true, b.v,
                                                      acc[j], false, false);
    }
  }

  float sA = fmaxf(amax[ai] * (1.0f / 127.0f), 1e-8f);
  float sW = fmaxf(amax[wi] * (1.0f / 127.0f), 1e-8f);
  float s = sA * sW;

  #pragma unroll
  for (int j = 0; j < 4; ++j)
    #pragma unroll
    for (int r = 0; r < 8; ++r) {
      int m = mTile + r + 8 * hl;
      int n = nBase + j * 16 + l16;
      float v = (float)acc[j][r] * s + bias[n];
      if (mode == 0) {
        int which = n >> 10;
        int cin = n & (C_ - 1);
        int h = cin >> 6, d = cin & 63;
        int bb = m >> 11, t = m & (T_ - 1);
        _Float16* qkv = (_Float16*)out_p;
        qkv[((((size_t)which * B_ + bb) * H_ + h) * T_ + t) * D_ + d] =
            (_Float16)v;
      } else {
        ((float*)out_p)[(size_t)m * N + n] = v;
      }
    }
}

#define AW 4
__global__ __launch_bounds__(AW * 32) void
attn_kernel(const _Float16* __restrict__ qkv, float* __restrict__ aout,
            float* __restrict__ amax_slot) {
  __shared__ _Float16 smem[AW][16 * 32 + 32 * 64 + 64 * 32];
  int wib  = threadIdx.x >> 5;
  int lane = threadIdx.x & 31;
  int gw = blockIdx.x * AW + wib;
  int qt = gw & 127;
  int bh = gw >> 7;
  if (bh >= B_ * H_) return;
  int hl = lane >> 4, l16 = lane & 15;

  const _Float16* qb = qkv + ((size_t)(0 * B_ * H_ + bh)) * T_ * D_;
  const _Float16* kb = qkv + ((size_t)(1 * B_ * H_ + bh)) * T_ * D_;
  const _Float16* vb = qkv + ((size_t)(2 * B_ * H_ + bh)) * T_ * D_;
  _Float16* pL   = smem[wib];
  _Float16* vRow = smem[wib] + 16 * 32;
  _Float16* vT   = smem[wib] + 16 * 32 + 32 * 64;

#ifdef HAVE_TDM
  unsigned lds_off = __builtin_amdgcn_readfirstlane(
      (unsigned)(unsigned long long)(void*)vRow);
#endif

  HFrag aq[2];
  const _Float16* qrow = qb + ((size_t)qt * 16 + l16) * D_;
  #pragma unroll
  for (int c2 = 0; c2 < 2; ++c2) {
    aq[c2].q[0] = *(const u32x4*)(qrow + c2 * 32 + hl * 8);
    aq[c2].q[1] = *(const u32x4*)(qrow + c2 * 32 + 16 + hl * 8);
  }

  v8f o[4], mrun, lrun;
  #pragma unroll
  for (int r = 0; r < 8; ++r) { mrun[r] = -INFINITY; lrun[r] = 0.0f; }
  #pragma unroll
  for (int db = 0; db < 4; ++db)
    #pragma unroll
    for (int r = 0; r < 8; ++r) o[db][r] = 0.0f;

  const float scale = 0.125f;
  int kend = qt * 16 + 16;

  for (int k0 = 0; k0 < kend; k0 += 32) {
#ifdef HAVE_TDM
    {
      unsigned long long ga =
          (unsigned long long)(const void*)(vb + (size_t)k0 * D_);
      unsigned ga_lo = __builtin_amdgcn_readfirstlane((unsigned)ga);
      unsigned ga_hi = __builtin_amdgcn_readfirstlane((unsigned)(ga >> 32));
      u32x4 g0;
      g0[0] = 1u;
      g0[1] = lds_off;
      g0[2] = ga_lo;
      g0[3] = (ga_hi & 0x01FFFFFFu) | (2u << 30);
      v8i g1;
      g1[0] = 0x00010000;
      g1[1] = (int)(64u << 16);
      g1[2] = (int)(2048u << 16);
      g1[3] = (int)(64u << 16);
      g1[4] = 32;
      g1[5] = 64;
      g1[6] = 0; g1[7] = 0;
      i32x4 gz; gz[0] = gz[1] = gz[2] = gz[3] = 0;
#if __clang_major__ >= 23
      v8i gz8;
      #pragma unroll
      for (int i = 0; i < 8; ++i) gz8[i] = 0;
      __builtin_amdgcn_tensor_load_to_lds(g0, g1, gz, gz, gz8, 0);
#else
      __builtin_amdgcn_tensor_load_to_lds(g0, g1, gz, gz, 0);
#endif
    }
#else
    {
      const _Float16* vrow_g = vb + (size_t)(k0 + lane) * D_;
      #pragma unroll
      for (int c = 0; c < 8; ++c)
        *(u32x4*)(vRow + lane * 64 + c * 8) = *(const u32x4*)(vrow_g + c * 8);
    }
#endif

    v8f s[2];
    #pragma unroll
    for (int t = 0; t < 2; ++t) {
      v8f sacc;
      #pragma unroll
      for (int r = 0; r < 8; ++r) sacc[r] = 0.0f;
      const _Float16* krow = kb + (size_t)(k0 + t * 16 + l16) * D_;
      __builtin_prefetch(krow + 32 * D_, 0, 3);
      #pragma unroll
      for (int c2 = 0; c2 < 2; ++c2) {
        HFrag bf;
        bf.q[0] = *(const u32x4*)(krow + c2 * 32 + hl * 8);
        bf.q[1] = *(const u32x4*)(krow + c2 * 32 + 16 + hl * 8);
        sacc = __builtin_amdgcn_wmma_f32_16x16x32_f16(
            false, aq[c2].v, false, bf.v, (short)0, sacc, false, false);
      }
      s[t] = sacc;
    }
    #pragma unroll
    for (int t = 0; t < 2; ++t)
      #pragma unroll
      for (int r = 0; r < 8; ++r) {
        int kidx = k0 + t * 16 + l16;
        int qrw = qt * 16 + r + 8 * hl;
        float sv = s[t][r] * scale;
        s[t][r] = (kidx <= qrw) ? sv : -3.0e38f;
      }
    v8f rmax, rs;
    #pragma unroll
    for (int r = 0; r < 8; ++r) rmax[r] = fmaxf(s[0][r], s[1][r]);
    #pragma unroll
    for (int mk = 1; mk < 16; mk <<= 1)
      #pragma unroll
      for (int r = 0; r < 8; ++r)
        rmax[r] = fmaxf(rmax[r], __shfl_xor(rmax[r], mk, 32));
    #pragma unroll
    for (int r = 0; r < 8; ++r) {
      float mnew = fmaxf(mrun[r], rmax[r]);
      float p0 = __expf(s[0][r] - mnew);
      float p1 = __expf(s[1][r] - mnew);
      s[0][r] = p0; s[1][r] = p1;
      rs[r] = p0 + p1;
      float esc = __expf(mrun[r] - mnew);
      mrun[r] = mnew;
      lrun[r] *= esc;
      #pragma unroll
      for (int db = 0; db < 4; ++db) o[db][r] *= esc;
    }
    #pragma unroll
    for (int mk = 1; mk < 16; mk <<= 1)
      #pragma unroll
      for (int r = 0; r < 8; ++r) rs[r] += __shfl_xor(rs[r], mk, 32);
    #pragma unroll
    for (int r = 0; r < 8; ++r) lrun[r] += rs[r];

    #pragma unroll
    for (int t = 0; t < 2; ++t)
      #pragma unroll
      for (int r = 0; r < 8; ++r)
        pL[(r + 8 * hl) * 32 + t * 16 + l16] = (_Float16)s[t][r];

#ifdef HAVE_TDM
    wait_tensor0();
#endif
    wave_lds_sync();
    #pragma unroll
    for (int c = 0; c < 8; ++c) {
      u32x4 chunk = *(const u32x4*)(vRow + lane * 64 + c * 8);
      _Float16 tmp[8];
      *(u32x4*)tmp = chunk;
      #pragma unroll
      for (int j = 0; j < 8; ++j) vT[(c * 8 + j) * 32 + lane] = tmp[j];
    }
    wave_lds_sync();

    HFrag ap;
    ap.q[0] = *(const u32x4*)(pL + l16 * 32 + hl * 8);
    ap.q[1] = *(const u32x4*)(pL + l16 * 32 + 16 + hl * 8);
    #pragma unroll
    for (int db = 0; db < 4; ++db) {
      HFrag bf;
      const _Float16* vcol = vT + (db * 16 + l16) * 32 + hl * 8;
      bf.q[0] = *(const u32x4*)(vcol);
      bf.q[1] = *(const u32x4*)(vcol + 16);
      o[db] = __builtin_amdgcn_wmma_f32_16x16x32_f16(
          false, ap.v, false, bf.v, (short)0, o[db], false, false);
    }
    wave_lds_sync();
  }

  int bb = bh >> 4, h = bh & 15;
  float la = 0.0f;
  #pragma unroll
  for (int db = 0; db < 4; ++db)
    #pragma unroll
    for (int r = 0; r < 8; ++r) {
      int m = qt * 16 + r + 8 * hl;
      int d = db * 16 + l16;
      float val = o[db][r] / lrun[r];
      aout[((size_t)(bb * T_ + m)) * C_ + h * 64 + d] = val;
      la = fmaxf(la, fabsf(val));
    }
  #pragma unroll
  for (int mk = 1; mk < 32; mk <<= 1) la = fmaxf(la, __shfl_xor(la, mk, 32));
  if (lane == 0) atomicMax((unsigned int*)amax_slot, __float_as_uint(la));
}

extern "C" void kernel_launch(void* const* d_in, const int* in_sizes, int n_in,
                              void* d_out, int out_size, void* d_ws, size_t ws_size,
                              hipStream_t stream) {
  (void)in_sizes; (void)n_in; (void)out_size; (void)ws_size;
  const float* x     = (const float*)d_in[0];
  const float* Wqkv  = (const float*)d_in[1];
  const float* bqkv  = (const float*)d_in[2];
  const float* Wproj = (const float*)d_in[3];
  const float* bproj = (const float*)d_in[4];
  float* out = (float*)d_out;

  char* ws = (char*)d_ws;
  float*       amax  = (float*)(ws + OFF_AMAX);
  signed char* xq    = (signed char*)(ws + OFF_XQ);
  signed char* wqkvq = (signed char*)(ws + OFF_WQKV);
  signed char* wprjq = (signed char*)(ws + OFF_WPROJ);
  _Float16*    qkv   = (_Float16*)(ws + OFF_QKV);
  float*       aout  = (float*)(ws + OFF_AOUT);
  signed char* outq  = (signed char*)(ws + OFF_OUTQ);

  init_amax_kernel<<<1, 64, 0, stream>>>(amax);

  amax_kernel<<<512, 256, 0, stream>>>(x,     (BT * C_) / 4, amax + 0);
  amax_kernel<<<512, 256, 0, stream>>>(Wqkv,  (C3 * C_) / 4, amax + 1);
  amax_kernel<<<256, 256, 0, stream>>>(Wproj, (C_ * C_) / 4, amax + 2);

  quant_kernel<<<512, 256, 0, stream>>>(x,     (BT * C_) / 4, amax + 0, xq);
  quant_kernel<<<512, 256, 0, stream>>>(Wqkv,  (C3 * C_) / 4, amax + 1, wqkvq);
  quant_kernel<<<256, 256, 0, stream>>>(Wproj, (C_ * C_) / 4, amax + 2, wprjq);

  {
    int waves = (BT / 16) * (C3 / 64);
    gemm_iu8_kernel<<<waves / 8, 256, 0, stream>>>(
        xq, wqkvq, bqkv, amax, 0, 1, BT, C3, C_, 0, (void*)qkv);
  }

  attn_kernel<<<(B_ * H_ * 128) / AW, AW * 32, 0, stream>>>(qkv, aout, amax + 3);

  quant_kernel<<<512, 256, 0, stream>>>(aout, (BT * C_) / 4, amax + 3, outq);

  {
    int waves = (BT / 16) * (C_ / 64);
    gemm_iu8_kernel<<<waves / 8, 256, 0, stream>>>(
        outq, wprjq, bproj, amax, 3, 2, BT, C_, C_, 1, (void*)out);
  }
}
